// CollaborativeAttention_59657095741492
// MI455X (gfx1250) — matrix operand order repaired and checked by reading, not compiled, not run on hardware
//
#include <hip/hip_runtime.h>
#include <hip/hip_bf16.h>
#include <math.h>

// ---------------------------------------------------------------------------
// CollaborativeAttention for MI455X (gfx1250): bf16 WMMA flash-attention
// Software-pipelined loads (double-buffered register tiles) so the compiler
// can emit partial s_wait_loadcnt instead of full drains before each v_wmma.
// ---------------------------------------------------------------------------

typedef __attribute__((ext_vector_type(16))) __bf16 v16bf;
typedef __attribute__((ext_vector_type(8)))  float  v8f;

#define B_  16
#define S_  512
#define D_  512
#define H_  8
#define M_  (B_ * S_)      // 8192
#define NEG_BIG (-1.0e30f)

union ABf { v16bf v; unsigned short h[16]; uint4 q[2]; };
union VF  { v8f v; float f[8]; };

static __device__ __forceinline__ unsigned short f2bf(float f) {
  unsigned int x = __float_as_uint(f);
  unsigned int r = x + 0x7fffu + ((x >> 16) & 1u);   // round-to-nearest-even
  return (unsigned short)(r >> 16);
}
static __device__ __forceinline__ float bf2f(unsigned short h) {
  return __uint_as_float(((unsigned int)h) << 16);
}
static __device__ __forceinline__ v8f wmma_bf16(v16bf a, v16bf b, v8f c) {
  // D(16x16 f32) = A(16x32 bf16) * B(32x16 bf16) + C
  return __builtin_amdgcn_wmma_f32_16x16x32_bf16(false, a, false, b,
                                                 (short)0, c, false, false);
}

// ---------------------------------------------------------------------------
// fp32 -> bf16 conversion
__global__ void k_cvt_f32_bf16(const float* __restrict__ in,
                               unsigned short* __restrict__ out, int n) {
  int i = blockIdx.x * blockDim.x + threadIdx.x;
  if (i < n) out[i] = f2bf(in[i]);
}

// W [N=512, K=512] fp32 -> Wt [N, K] bf16 (B-matrix friendly layout)
__global__ void k_cvt_wt(const float* __restrict__ W,
                         unsigned short* __restrict__ Wt) {
  int i = blockIdx.x * blockDim.x + threadIdx.x;   // i = k*512 + n
  int k = i >> 9, n = i & 511;
  Wt[n * 512 + k] = f2bf(W[n * 512 + k]);
}

// ---------------------------------------------------------------------------
// Generic bf16 WMMA GEMM: C[8192,512] = A[8192,512] @ Bm[512,512]
// Double-buffered register tiles: iteration kc+1's loads issue before the
// WMMAs of iteration kc (in-order VMEM returns -> partial loadcnt waits).
// mode 0: store bf16 row-major (q)
// mode 1: store bf16 row-major                                        (k)
// mode 2: store bf16 transposed per-batch + bias[n]  vT[(b*512 + n)*512 + t]  (v)
// mode 3: store fp32  acc + bias[n] + resid[m*512+n]                  (dense)
__global__ __launch_bounds__(256)
void k_gemm_bf16(const unsigned short* __restrict__ A,
                 const unsigned short* __restrict__ Bm,
                 int mode,
                 unsigned short* __restrict__ outh,
                 float* __restrict__ outf,
                 const float* __restrict__ bias,
                 const float* __restrict__ resid) {
  const int K = 512, N = 512;
  int gw   = (blockIdx.x * blockDim.x + threadIdx.x) >> 5;
  int lane = threadIdx.x & 31;
  int rowt = gw >> 3;                    // 512 row tiles of 16
  int colb = gw & 7;                     // 8 col blocks of 64
  int m0 = rowt * 16, n0 = colb * 64;
  int nlo = lane & 15, hi = lane >> 4;

  VF acc[4];
  #pragma unroll
  for (int t = 0; t < 4; t++)
    #pragma unroll
    for (int r = 0; r < 8; r++) acc[t].f[r] = 0.0f;

  const unsigned short* arow = A + (m0 + nlo) * K;
  const unsigned short* brow = Bm + (n0 + nlo) * K + hi * 8;   // + kc*32 per chunk

  ABf a[2];                 // 16-bit A layout (ISA 7.12.2)
  ABf bt[2][4];             // B layout: lane = N col, elems = K
  // prologue: chunk 0
  a[0].q[0] = *(const uint4*)(arow + hi * 8);
  a[0].q[1] = *(const uint4*)(arow + hi * 8 + 16);
  #pragma unroll
  for (int t = 0; t < 4; t++) {
    bt[0][t].q[0] = *(const uint4*)(brow + t * 16 * K);
    bt[0][t].q[1] = *(const uint4*)(brow + t * 16 * K + 16);
  }

  #pragma unroll
  for (int kc = 0; kc < 16; ++kc) {
    int cur = kc & 1, nxt = cur ^ 1;
    if (kc < 15) {                       // prefetch next chunk first
      int off = (kc + 1) * 32 + hi * 8;
      a[nxt].q[0] = *(const uint4*)(arow + off);
      a[nxt].q[1] = *(const uint4*)(arow + off + 16);
      const unsigned short* bp = brow + (kc + 1) * 32;
      #pragma unroll
      for (int t = 0; t < 4; t++) {
        bt[nxt][t].q[0] = *(const uint4*)(bp + t * 16 * K);
        bt[nxt][t].q[1] = *(const uint4*)(bp + t * 16 * K + 16);
      }
    }
    #pragma unroll
    for (int t = 0; t < 4; t++)
      acc[t].v = wmma_bf16(a[cur].v, bt[cur][t].v, acc[t].v);
  }

  #pragma unroll
  for (int t = 0; t < 4; t++) {
    #pragma unroll
    for (int r = 0; r < 8; r++) {
      int m = m0 + r + 8 * hi;           // C layout: lane=N, vgpr r = row r/r+8
      int n = n0 + t * 16 + nlo;
      float vfl = acc[t].f[r];
      if (mode == 0) {
        outh[m * N + n] = f2bf(vfl);
      } else if (mode == 1) {
        outh[m * N + n] = f2bf(vfl);
      } else if (mode == 2) {
        int bb = m >> 9, tt = m & 511;
        outh[((bb << 9) + n) * 512 + tt] = f2bf(vfl + bias[n]);
      } else {
        outf[m * N + n] = vfl + bias[n] + resid[m * N + n];
      }
    }
  }
}

// ---------------------------------------------------------------------------
// content bias: cb[m*8+h] = dot(x[m,:], Wcb[h,:])
__global__ void k_cb(const float* __restrict__ x, const float* __restrict__ Wcb,
                     float* __restrict__ cb) {
  int i = blockIdx.x * blockDim.x + threadIdx.x;   // m*8 + h
  int m = i >> 3, h = i & 7;
  const float* xr = x + m * 512;
  const float* wr = Wcb + h * 512;
  float acc = 0.f;
  #pragma unroll 8
  for (int k = 0; k < 512; ++k) acc = fmaf(xr[k], wr[k], acc);
  cb[i] = acc;
}

// temporal bias MLP: 1 -> 64 (leaky 0.2) -> 1 on u = 1/log(e + T)
__global__ void k_tbias(const float* __restrict__ tm,
                        const float* __restrict__ w1, const float* __restrict__ b1,
                        const float* __restrict__ w2, const float* __restrict__ b2,
                        float* __restrict__ out, int n) {
  __shared__ float sw1[64], sb1[64], sw2[64];
  int tid = threadIdx.x;
  if (tid < 64) { sw1[tid] = w1[tid]; sb1[tid] = b1[tid]; sw2[tid] = w2[tid]; }
  __syncthreads();
  int i = blockIdx.x * blockDim.x + tid;
  if (i >= n) return;
  float u = 1.0f / __logf(2.71828182845904523536f + tm[i]);
  float acc = b2[0];
  #pragma unroll 8
  for (int j = 0; j < 64; ++j) {
    float hj = fmaf(u, sw1[j], sb1[j]);
    hj = (hj > 0.f) ? hj : 0.2f * hj;
    acc = fmaf(hj, sw2[j], acc);
  }
  out[i] = acc;
}

// ---------------------------------------------------------------------------
// Flash attention: one wave per (b, h, 16-query tile).
//   scores = (mixed_q @ k^T + tbias + cb) masked, *0.125, online softmax
//   ctx    = probs @ v   (probs transposed C->A layout through LDS)
__global__ __launch_bounds__(128, 1)
void k_attn(const unsigned short* __restrict__ qh,
            const unsigned short* __restrict__ khT,   // [B*S, DK] bf16
            const unsigned short* __restrict__ vh,    // [B, DV, S] bf16
            const float* __restrict__ mixing,         // [H, DK] fp32
            const float* __restrict__ cb,             // [B*S, H] fp32
            const float* __restrict__ tb,             // [B, S, S] fp32
            unsigned short* __restrict__ ctxh) {      // [B*S, DV] bf16
  __shared__ unsigned short plds[4 * 512];            // 16x32 bf16 per wave
  int wid  = threadIdx.x >> 5;
  int lane = threadIdx.x & 31;
  int gw   = blockIdx.x * 4 + wid;
  int b    = gw >> 8;
  int rem  = gw & 255;
  int h    = rem >> 5;
  int q0   = (rem & 31) << 4;
  int nlo  = lane & 15, hi = lane >> 4;
  unsigned short* myp = plds + wid * 512;

  // Resident mixed-q A-tiles: 16 chunks of K=32 (128 VGPRs)
  ABf aq[16];
  {
    const unsigned short* qrow = qh + ((b << 9) + q0 + nlo) * 512;
    #pragma unroll
    for (int kc = 0; kc < 16; ++kc) {
      int off = kc * 32 + hi * 8;
      aq[kc].q[0] = *(const uint4*)(qrow + off);
      aq[kc].q[1] = *(const uint4*)(qrow + off + 16);
      #pragma unroll
      for (int j = 0; j < 16; ++j) {
        int kk = kc * 32 + hi * 8 + (j & 7) + ((j >> 3) << 4);
        float f = bf2f(aq[kc].h[j]) * mixing[h * 512 + kk];
        aq[kc].h[j] = f2bf(f);
      }
    }
  }

  VF cacc[4];
  #pragma unroll
  for (int t = 0; t < 4; t++)
    #pragma unroll
    for (int r = 0; r < 8; r++) cacc[t].f[r] = 0.f;
  float mrow[8], lrow[8];
  #pragma unroll
  for (int r = 0; r < 8; r++) { mrow[r] = NEG_BIG; lrow[r] = 0.f; }

  // causal early-exit; query row 0 (global) sees all keys (add_cls)
  int npairs = (q0 == 0) ? 16 : ((q0 + 47) >> 5);
  const float scale = 0.125f;   // 1/sqrt(64)

  for (int jp = 0; jp < npairs; ++jp) {
    int key0 = jp * 32;
    const unsigned short* kpbase = khT + ((b << 9) + key0 + nlo) * 512 + hi * 8;
    const unsigned short* vp = vh + ((b << 9) + (h << 6) + nlo) * 512 + key0 + hi * 8;
    __builtin_prefetch(vp, 0, 1);        // global_prefetch: v rows for this pair

    VF s0, s1;
    #pragma unroll
    for (int r = 0; r < 8; r++) { s0.f[r] = 0.f; s1.f[r] = 0.f; }

    // double-buffered k^T B-tiles (two 16-byte loads per lane per tile, at k = 8h and 16 + 8h)
    ABf kb[2][2];
    kb[0][0].q[0] = *(const uint4*)(kpbase);
    kb[0][0].q[1] = *(const uint4*)(kpbase + 16);
    kb[0][1].q[0] = *(const uint4*)(kpbase + 16 * 512);
    kb[0][1].q[1] = *(const uint4*)(kpbase + 16 * 512 + 16);

    #pragma unroll
    for (int kc = 0; kc < 16; ++kc) {
      int cur = kc & 1, nxt = cur ^ 1;
      if (kc < 15) {                     // prefetch next feature chunk
        const unsigned short* kp = kpbase + (kc + 1) * 32;
        kb[nxt][0].q[0] = *(const uint4*)(kp);
        kb[nxt][0].q[1] = *(const uint4*)(kp + 16);
        kb[nxt][1].q[0] = *(const uint4*)(kp + 16 * 512);
        kb[nxt][1].q[1] = *(const uint4*)(kp + 16 * 512 + 16);
      }
      s0.v = wmma_bf16(aq[kc].v, kb[cur][0].v, s0.v);
      s1.v = wmma_bf16(aq[kc].v, kb[cur][1].v, s1.v);
    }

    // hoist v B-tiles: latency hides under the softmax VALU section below
    ABf vb[4];
    #pragma unroll
    for (int t = 0; t < 4; t++) {
      vb[t].q[0] = *(const uint4*)(vp + t * 16 * 512);
      vb[t].q[1] = *(const uint4*)(vp + t * 16 * 512 + 16);
    }

    float cb0 = cb[((b << 9) + key0 + nlo) * 8 + h];
    float cb1 = cb[((b << 9) + key0 + 16 + nlo) * 8 + h];

    #pragma unroll
    for (int r = 0; r < 8; r++) {
      int row = q0 + r + 8 * hi;
      const float* tbr = tb + (((b << 9) + row) << 9);
      float v0 = s0.f[r] + tbr[key0 + nlo] + cb0;
      float v1 = s1.f[r] + tbr[key0 + 16 + nlo] + cb1;
      if (row != 0) {
        if (key0 + nlo > row)      v0 = NEG_BIG;
        if (key0 + 16 + nlo > row) v1 = NEG_BIG;
      }
      v0 *= scale; v1 *= scale;

      float tmax = fmaxf(v0, v1);
      #pragma unroll
      for (int o = 8; o > 0; o >>= 1) tmax = fmaxf(tmax, __shfl_xor(tmax, o, 16));
      float newm = fmaxf(mrow[r], tmax);
      float corr = __expf(mrow[r] - newm);
      float p0 = __expf(v0 - newm);
      float p1 = __expf(v1 - newm);
      float ps = p0 + p1;
      #pragma unroll
      for (int o = 8; o > 0; o >>= 1) ps += __shfl_xor(ps, o, 16);
      lrow[r] = lrow[r] * corr + ps;
      mrow[r] = newm;
      #pragma unroll
      for (int t = 0; t < 4; t++) cacc[t].f[r] *= corr;

      int prow = r + 8 * hi;      // probs bf16 -> LDS (row-major 16x32)
      myp[prow * 32 + nlo]      = f2bf(p0);
      myp[prow * 32 + 16 + nlo] = f2bf(p1);
    }

    asm volatile("s_wait_dscnt 0" ::: "memory");   // same-wave LDS RAW

    ABf pa;   // read probs back in 16-bit A layout
    {
      const unsigned short* pr = myp + nlo * 32 + hi * 8;
      pa.q[0] = *(const uint4*)(pr);
      pa.q[1] = *(const uint4*)(pr + 16);
    }

    #pragma unroll
    for (int t = 0; t < 4; t++)
      cacc[t].v = wmma_bf16(pa.v, vb[t].v, cacc[t].v);
  }

  #pragma unroll
  for (int r = 0; r < 8; r++) {
    float inv = 1.0f / lrow[r];
    int row = (b << 9) + q0 + r + 8 * hi;
    #pragma unroll
    for (int t = 0; t < 4; t++)
      ctxh[row * 512 + (h << 6) + t * 16 + nlo] = f2bf(cacc[t].f[r] * inv);
  }
}

// ---------------------------------------------------------------------------
// In-place LayerNorm over rows of 512
__global__ void k_ln(float* __restrict__ y, const float* __restrict__ g,
                     const float* __restrict__ be) {
  __shared__ float ssum[256], ssq[256];
  int m = blockIdx.x, tid = threadIdx.x;
  float v0 = y[m * 512 + tid];
  float v1 = y[m * 512 + tid + 256];
  ssum[tid] = v0 + v1;
  ssq[tid]  = v0 * v0 + v1 * v1;
  __syncthreads();
  for (int o = 128; o > 0; o >>= 1) {
    if (tid < o) { ssum[tid] += ssum[tid + o]; ssq[tid] += ssq[tid + o]; }
    __syncthreads();
  }
  float mu  = ssum[0] * (1.0f / 512.0f);
  float var = ssq[0] * (1.0f / 512.0f) - mu * mu;
  float rs  = rsqrtf(var + 1e-5f);
  y[m * 512 + tid]       = (v0 - mu) * rs * g[tid]       + be[tid];
  y[m * 512 + tid + 256] = (v1 - mu) * rs * g[tid + 256] + be[tid + 256];
}

// ---------------------------------------------------------------------------
extern "C" void kernel_launch(void* const* d_in, const int* in_sizes, int n_in,
                              void* d_out, int out_size, void* d_ws, size_t ws_size,
                              hipStream_t stream) {
  const float* x    = (const float*)d_in[0];
  const float* tm   = (const float*)d_in[1];
  const float* Wq   = (const float*)d_in[2];
  const float* Wk   = (const float*)d_in[3];
  const float* Wcb  = (const float*)d_in[4];
  const float* Wv   = (const float*)d_in[5];
  const float* bv   = (const float*)d_in[6];
  const float* mixg = (const float*)d_in[7];
  const float* tb1w = (const float*)d_in[8];
  const float* tb1b = (const float*)d_in[9];
  const float* tb2w = (const float*)d_in[10];
  const float* tb2b = (const float*)d_in[11];
  const float* Wd   = (const float*)d_in[12];
  const float* bd   = (const float*)d_in[13];
  const float* lng  = (const float*)d_in[14];
  const float* lnb  = (const float*)d_in[15];
  float* out = (float*)d_out;

  char* ws = (char*)d_ws;
  size_t off = 0;
  auto alloc = [&](size_t bytes) -> char* {
    char* p = ws + off;
    off += (bytes + 255) & ~(size_t)255;
    return p;
  };
  unsigned short* xh   = (unsigned short*)alloc((size_t)M_ * 512 * 2);
  unsigned short* qh   = (unsigned short*)alloc((size_t)M_ * 512 * 2);
  unsigned short* khT  = (unsigned short*)alloc((size_t)M_ * 512 * 2);
  unsigned short* vh   = (unsigned short*)alloc((size_t)M_ * 512 * 2);
  unsigned short* ctxh = (unsigned short*)alloc((size_t)M_ * 512 * 2);
  unsigned short* WqT  = (unsigned short*)alloc(512 * 512 * 2);
  unsigned short* WkT  = (unsigned short*)alloc(512 * 512 * 2);
  unsigned short* WvT  = (unsigned short*)alloc(512 * 512 * 2);
  unsigned short* WdT  = (unsigned short*)alloc(512 * 512 * 2);
  float* cbw = (float*)alloc((size_t)M_ * H_ * 4);
  float* tbw = (float*)alloc((size_t)B_ * S_ * S_ * 4);

  // stage bf16 operands
  k_cvt_f32_bf16<<<(M_ * 512 + 255) / 256, 256, 0, stream>>>(x, xh, M_ * 512);
  k_cvt_wt<<<1024, 256, 0, stream>>>(Wq, WqT);
  k_cvt_wt<<<1024, 256, 0, stream>>>(Wk, WkT);
  k_cvt_wt<<<1024, 256, 0, stream>>>(Wv, WvT);
  k_cvt_wt<<<1024, 256, 0, stream>>>(Wd, WdT);

  // projections (WMMA): 4096 waves, 8 wave32 per block
  k_gemm_bf16<<<512, 256, 0, stream>>>(xh, WqT, 0, qh, nullptr, nullptr, nullptr);
  k_gemm_bf16<<<512, 256, 0, stream>>>(xh, WkT, 1, khT, nullptr, nullptr, nullptr);
  k_gemm_bf16<<<512, 256, 0, stream>>>(xh, WvT, 2, vh, nullptr, bv, nullptr);

  // biases
  k_cb<<<(M_ * H_ + 255) / 256, 256, 0, stream>>>(x, Wcb, cbw);
  k_tbias<<<(B_ * S_ * S_ + 255) / 256, 256, 0, stream>>>(tm, tb1w, tb1b, tb2w,
                                                          tb2b, tbw, B_ * S_ * S_);

  // flash attention (WMMA)
  k_attn<<<1024, 128, 0, stream>>>(qh, khT, vh, mixg, cbw, tbw, ctxh);

  // dense + residual (WMMA), then LayerNorm in place on d_out
  k_gemm_bf16<<<512, 256, 0, stream>>>(ctxh, WdT, 3, nullptr, out, bd, x);
  k_ln<<<M_, 256, 0, stream>>>(out, lng, lnb);
}
